// PairBiasSelfAttention_11854109737243
// MI455X (gfx1250) — matrix operand order repaired and checked by reading, not compiled, not run on hardware
//
#include <hip/hip_runtime.h>
#include <hip/hip_bf16.h>

// ---------------------------------------------------------------------------
// PairBias Self-Attention for MI455X (gfx1250, wave32)
//   - all GEMMs on v_wmma_f32_16x16x32_f16
//   - A-tiles staged to LDS via TDM tensor_load_to_lds (s_wait_tensorcnt)
//   - B-stream register double-buffered against the WMMAs
// ---------------------------------------------------------------------------

typedef _Float16 f16;
typedef __attribute__((ext_vector_type(16))) _Float16 v16h;
typedef __attribute__((ext_vector_type(8)))  _Float16 v8h;
typedef __attribute__((ext_vector_type(8)))  float    v8f;
typedef unsigned int u32;
typedef unsigned long long u64;
typedef __attribute__((ext_vector_type(4))) u32 u32x4;
typedef __attribute__((ext_vector_type(8))) int i32x8;
typedef __attribute__((ext_vector_type(4))) int i32x4;

#define L_DIM   1024
#define B_DIM   2
#define E_DIM   512
#define H_NUM   8
#define D_HEAD  64
#define P_DIM   64
#define ROWS    (L_DIM * B_DIM)        // 2048 flattened (l,b) rows

// ---- WMMA fragment helpers (wave32 layouts per CDNA5 ISA 7.12.2) ----------

// A fragment: 16x32 f16 rows. lane<16: K=k0+[0..7],k0+[16..23];
// lane>=16: K=k0+[8..15],k0+[24..31].
__device__ __forceinline__ v16h frag_a(const f16* base, int lda, int k0, int lane) {
  const int m  = lane & 15;
  const int kh = lane >> 4;
  const f16* p = base + m * lda + k0 + kh * 8;
  v8h lo = *(const v8h*)(p);
  v8h hi = *(const v8h*)(p + 16);
  v16h a;
#pragma unroll
  for (int i = 0; i < 8; ++i) { a[i] = lo[i]; a[8 + i] = hi[i]; }
  return a;
}

// B fragment: 32x16 where B[k][n] = Wrow[n][k]; lane<16: K=k0+[0..7],k0+[16..23];
// lane>=16: K=k0+[8..15],k0+[24..31].
__device__ __forceinline__ v16h frag_b(const f16* base, int ldb, int k0, int lane) {
  const int n  = lane & 15;
  const int kh = lane >> 4;
  const f16* p = base + n * ldb + k0 + kh * 8;
  v8h lo = *(const v8h*)(p);
  v8h hi = *(const v8h*)(p + 16);
  v16h b;
#pragma unroll
  for (int i = 0; i < 8; ++i) { b[i] = lo[i]; b[8 + i] = hi[i]; }
  return b;
}

__device__ __forceinline__ v8f wmma_f16f32(v16h a, v16h b, v8f c) {
  return __builtin_amdgcn_wmma_f32_16x16x32_f16(false, a, false, b,
                                                (short)0, c, false, false);
}

// ---- TDM: stage a 16-row x d0-elem f16 tile (row stride `stride`) to LDS --
// D# per CDNA5 ISA ch.8: group0 = {count/type/lds_addr/global_addr},
// group1 = {data_size, tensor dims, tile dims, dim0 stride}.
__device__ __forceinline__ void stage_a_tile(f16* sA, const f16* g,
                                             int d0, int rows, int stride,
                                             int lane) {
#if __has_builtin(__builtin_amdgcn_tensor_load_to_lds)
  u64 ga = (u64)(const void*)g;
  u32x4 g0;
  g0[0] = 1u;                                           // count=1, user mode
  g0[1] = (u32)(size_t)sA;                              // lds_addr (low 32 = LDS off)
  g0[2] = (u32)ga;                                      // global_addr[31:0]
  g0[3] = ((u32)(ga >> 32) & 0x01FFFFFFu) | (2u << 30); // addr[56:32] | type=2
  i32x8 g1;
  g1[0] = 1 << 16;                                      // data_size=1 -> 2 bytes
  g1[1] = (int)(((u32)d0 & 0xFFFFu) << 16);             // tensor_dim0[15:0]
  g1[2] = (int)(((u32)d0 >> 16) | (((u32)rows & 0xFFFFu) << 16)); // dim0 hi|dim1 lo
  g1[3] = (int)(((u32)rows >> 16) | ((u32)d0 << 16));   // dim1 hi | tile_dim0
  g1[4] = rows & 0xFFFF;                                // tile_dim1 (tile_dim2=0)
  g1[5] = stride;                                       // tensor_dim0_stride[31:0]
  g1[6] = 0;
  g1[7] = 0;
  i32x4 z4 = {0, 0, 0, 0};
  asm volatile("" ::: "memory");
#if defined(__clang_major__) && (__clang_major__ >= 23)
  i32x8 z8 = {0, 0, 0, 0, 0, 0, 0, 0};
  __builtin_amdgcn_tensor_load_to_lds(g0, g1, z4, z4, z8, 0);
#else
  __builtin_amdgcn_tensor_load_to_lds(g0, g1, z4, z4, 0);
#endif
  __builtin_amdgcn_s_wait_tensorcnt(0);
  asm volatile("" ::: "memory");
#else
  const int vpr = d0 >> 3;                              // v8h chunks per row
  for (int i = lane; i < rows * vpr; i += 32) {
    int r = i / vpr, c = i - r * vpr;
    ((v8h*)sA)[i] = *(const v8h*)(g + r * stride + c * 8);
  }
#endif
}

// ---- software-pipelined 16x64 GEMM tile: A (often LDS) x 4 B sub-tiles ----
__device__ __forceinline__ void gemm_16x64(const f16* A, int lda,
                                           const f16* B, int ldb, int ktot,
                                           int lane, v8f acc[4]) {
  v16h a_c  = frag_a(A, lda, 0, lane);
  v16h b_c0 = frag_b(B,            ldb, 0, lane);
  v16h b_c1 = frag_b(B + 16 * ldb, ldb, 0, lane);
  v16h b_c2 = frag_b(B + 32 * ldb, ldb, 0, lane);
  v16h b_c3 = frag_b(B + 48 * ldb, ldb, 0, lane);
  for (int k0 = 32; k0 < ktot; k0 += 32) {
    v16h a_n  = frag_a(A, lda, k0, lane);
    v16h b_n0 = frag_b(B,            ldb, k0, lane);
    v16h b_n1 = frag_b(B + 16 * ldb, ldb, k0, lane);
    v16h b_n2 = frag_b(B + 32 * ldb, ldb, k0, lane);
    v16h b_n3 = frag_b(B + 48 * ldb, ldb, k0, lane);
    acc[0] = wmma_f16f32(a_c, b_c0, acc[0]);
    acc[1] = wmma_f16f32(a_c, b_c1, acc[1]);
    acc[2] = wmma_f16f32(a_c, b_c2, acc[2]);
    acc[3] = wmma_f16f32(a_c, b_c3, acc[3]);
    a_c = a_n; b_c0 = b_n0; b_c1 = b_n1; b_c2 = b_n2; b_c3 = b_n3;
  }
  acc[0] = wmma_f16f32(a_c, b_c0, acc[0]);
  acc[1] = wmma_f16f32(a_c, b_c1, acc[1]);
  acc[2] = wmma_f16f32(a_c, b_c2, acc[2]);
  acc[3] = wmma_f16f32(a_c, b_c3, acc[3]);
}

__device__ __forceinline__ void zero_acc(v8f acc[4]) {
#pragma unroll
  for (int t = 0; t < 4; ++t)
#pragma unroll
    for (int r = 0; r < 8; ++r) acc[t][r] = 0.f;
}

// ---- K0: fp32 -> f16 weight conversion ------------------------------------
__global__ void pb_cvt_f16_kernel(const float* __restrict__ w,
                                  f16* __restrict__ o, int n) {
  int i = blockIdx.x * blockDim.x + threadIdx.x;
  if (i < n) o[i] = (f16)w[i];
}

// ---- K1: LayerNorm(query) -> xh f16 [2048,512] ----------------------------
__global__ void pb_ln_query_kernel(const float* __restrict__ q,
                                   const float* __restrict__ w,
                                   const float* __restrict__ b,
                                   f16* __restrict__ xh) {
  const int row  = (blockIdx.x * blockDim.x + threadIdx.x) >> 5;
  const int lane = threadIdx.x & 31;
  if (row >= ROWS) return;
  const float* rp = q + (size_t)row * E_DIM;
  float vals[16];
  float sum = 0.f, sq = 0.f;
#pragma unroll
  for (int i = 0; i < 16; ++i) {
    float v = rp[lane + i * 32];
    vals[i] = v; sum += v; sq += v * v;
  }
#pragma unroll
  for (int off = 16; off > 0; off >>= 1) {
    sum += __shfl_xor(sum, off, 32);
    sq  += __shfl_xor(sq,  off, 32);
  }
  const float mu   = sum * (1.f / E_DIM);
  const float var  = sq * (1.f / E_DIM) - mu * mu;
  const float rstd = rsqrtf(var + 1e-5f);
  f16* op = xh + (size_t)row * E_DIM;
#pragma unroll
  for (int i = 0; i < 16; ++i) {
    int j = lane + i * 32;
    op[j] = (f16)((vals[i] - mu) * rstd * w[j] + b[j]);
  }
}

// ---- K2: QKV projection GEMM (wave -> 16x64 tile; A staged via TDM) -------
__global__ void pb_qkv_gemm_kernel(const f16* __restrict__ xh,
                                   const f16* w0, const f16* w1, const f16* w2,
                                   f16* o0, f16* o1, f16* o2) {
  __shared__ __align__(128) f16 sA[16 * E_DIM];   // 16 KB
  const int lane = threadIdx.x;
  const int m0 = blockIdx.x * 16;
  const int n0 = blockIdx.y * 64;
  const int z  = blockIdx.z;
  const f16* W = (z == 0) ? w0 : ((z == 1) ? w1 : w2);
  f16*       O = (z == 0) ? o0 : ((z == 1) ? o1 : o2);
  const float scale = (z == 0) ? 0.125f : 1.0f;   // q * D^-0.5

  stage_a_tile(sA, xh + (size_t)m0 * E_DIM, E_DIM, 16, E_DIM, lane);

  v8f acc[4];
  zero_acc(acc);
  gemm_16x64(sA, E_DIM, W + (size_t)n0 * E_DIM, E_DIM, E_DIM, lane, acc);

  const int col = lane & 15, rsel = (lane >> 4) * 8;
#pragma unroll
  for (int t = 0; t < 4; ++t)
#pragma unroll
    for (int r = 0; r < 8; ++r)
      O[(size_t)(m0 + r + rsel) * E_DIM + n0 + t * 16 + col] =
          (f16)(acc[t][r] * scale);
}

// ---- K3: pair bias: LN(pairs) @ wp^T -> scores[B,H,L,S] (fp32) ------------
__global__ void pb_pair_bias_kernel(const float* __restrict__ pairs,
                                    const float* __restrict__ lnpw,
                                    const float* __restrict__ lnpb,
                                    const float* __restrict__ wp,
                                    float* __restrict__ scores) {
  __shared__ float s_wp[H_NUM * P_DIM];
  __shared__ float s_w[P_DIM];
  __shared__ float s_b[P_DIM];
  const int t = threadIdx.x;
  if (t < P_DIM) { s_w[t] = lnpw[t]; s_b[t] = lnpb[t]; }
  for (int i = t; i < H_NUM * P_DIM; i += blockDim.x) s_wp[i] = wp[i];
  __syncthreads();

  const size_t idx = (size_t)blockIdx.x * blockDim.x + t;  // over B*L*L
  const int b = (int)(idx >> 20);
  const int rem = (int)(idx & ((1u << 20) - 1));
  const int l = rem >> 10;
  const int s = rem & 1023;

  const float* p = pairs + idx * P_DIM;
  float y[P_DIM];
  float sum = 0.f, sq = 0.f;
#pragma unroll
  for (int i = 0; i < 16; ++i) {
    float4 f = ((const float4*)p)[i];
    y[4 * i + 0] = f.x; y[4 * i + 1] = f.y;
    y[4 * i + 2] = f.z; y[4 * i + 3] = f.w;
    sum += f.x + f.y + f.z + f.w;
    sq  += f.x * f.x + f.y * f.y + f.z * f.z + f.w * f.w;
  }
  const float mu   = sum * (1.f / P_DIM);
  const float var  = sq * (1.f / P_DIM) - mu * mu;
  const float rstd = rsqrtf(var + 1e-5f);
#pragma unroll
  for (int i = 0; i < P_DIM; ++i)
    y[i] = (y[i] - mu) * rstd * s_w[i] + s_b[i];

#pragma unroll
  for (int h = 0; h < H_NUM; ++h) {
    float acc = 0.f;
#pragma unroll
    for (int i = 0; i < P_DIM; ++i) acc += y[i] * s_wp[h * P_DIM + i];
    scores[((size_t)(b * H_NUM + h) << 20) + ((size_t)l << 10) + s] = acc;
  }
}

// ---- K4: scores += q @ k^T (wave -> 16x64 tile per (b,h)) -----------------
__global__ void pb_qk_kernel(const f16* __restrict__ qh,
                             const f16* __restrict__ kh,
                             float* __restrict__ scores) {
  __shared__ __align__(128) f16 sA[16 * D_HEAD];   // 2 KB
  const int lane = threadIdx.x;
  const int l0 = blockIdx.x * 16;
  const int s0 = blockIdx.y * 64;
  const int bh = blockIdx.z;
  const int b = bh >> 3, h = bh & 7;

  // q rows l0..l0+15 of head h: base + r*1024, 64 halves each
  stage_a_tile(sA, qh + (size_t)(l0 * B_DIM + b) * E_DIM + h * D_HEAD,
               D_HEAD, 16, B_DIM * E_DIM, lane);

  const f16* Bbase =
      kh + (size_t)(s0 * B_DIM + b) * E_DIM + h * D_HEAD;   // row s stride 1024

  v8f acc[4];
  zero_acc(acc);
  gemm_16x64(sA, D_HEAD, Bbase, B_DIM * E_DIM, D_HEAD, lane, acc);

  const int col = lane & 15, rsel = (lane >> 4) * 8;
  float* sp = scores + ((size_t)bh << 20);
#pragma unroll
  for (int t = 0; t < 4; ++t)
#pragma unroll
    for (int r = 0; r < 8; ++r) {
      size_t o = ((size_t)(l0 + r + rsel) << 10) + s0 + t * 16 + col;
      sp[o] += acc[t][r];   // add QK^T onto pre-written pair bias
    }
}

// ---- K5: row softmax -> probs f16 -----------------------------------------
__global__ void pb_softmax_kernel(const float* __restrict__ scores,
                                  f16* __restrict__ probs) {
  const int row  = (blockIdx.x * blockDim.x + threadIdx.x) >> 5;  // B*H*L rows
  const int lane = threadIdx.x & 31;
  const float* rp = scores + (size_t)row * L_DIM;
  float v[32];
  float mx = -3.0e38f;
#pragma unroll
  for (int i = 0; i < 32; ++i) {
    v[i] = rp[lane + i * 32];
    mx = fmaxf(mx, v[i]);
  }
#pragma unroll
  for (int off = 16; off > 0; off >>= 1)
    mx = fmaxf(mx, __shfl_xor(mx, off, 32));
  float sum = 0.f;
#pragma unroll
  for (int i = 0; i < 32; ++i) {
    v[i] = __expf(v[i] - mx);
    sum += v[i];
  }
#pragma unroll
  for (int off = 16; off > 0; off >>= 1)
    sum += __shfl_xor(sum, off, 32);
  const float inv = 1.f / sum;
  f16* op = probs + (size_t)row * L_DIM;
#pragma unroll
  for (int i = 0; i < 32; ++i)
    op[lane + i * 32] = (f16)(v[i] * inv);
}

// ---- K6: v [2048,512] -> vT [B,H,D,S] f16 ---------------------------------
__global__ void pb_vtrans_kernel(const f16* __restrict__ vh,
                                 f16* __restrict__ vT) {
  const int idx = blockIdx.x * blockDim.x + threadIdx.x;  // B*H*D*L = 1M
  const int s   = idx & 1023;
  int rem = idx >> 10;
  const int d  = rem & 63;
  const int bh = rem >> 6;
  const int b = bh >> 3, h = bh & 7;
  vT[idx] = vh[(size_t)(s * B_DIM + b) * E_DIM + h * D_HEAD + d];
}

// ---- K7: attn = probs @ v (wave -> 16 l-rows x 64 d per (b,h)) ------------
__global__ void pb_pv_kernel(const f16* __restrict__ probs,
                             const f16* __restrict__ vT,
                             f16* __restrict__ attn) {
  __shared__ __align__(128) f16 sA[16 * L_DIM];   // 32 KB
  const int lane = threadIdx.x;
  const int l0 = blockIdx.x * 16;
  const int bh = blockIdx.z;
  const int b = bh >> 3, h = bh & 7;

  stage_a_tile(sA, probs + ((size_t)bh << 20) + (size_t)l0 * L_DIM,
               L_DIM, 16, L_DIM, lane);

  const f16* Bbase = vT + (size_t)bh * D_HEAD * L_DIM;  // row d stride 1024

  v8f acc[4];
  zero_acc(acc);
  gemm_16x64(sA, L_DIM, Bbase, L_DIM, L_DIM, lane, acc);

  const int col = lane & 15, rsel = (lane >> 4) * 8;
#pragma unroll
  for (int t = 0; t < 4; ++t)
#pragma unroll
    for (int r = 0; r < 8; ++r)
      attn[(size_t)((l0 + r + rsel) * B_DIM + b) * E_DIM + h * D_HEAD +
           t * 16 + col] = (f16)acc[t][r];
}

// ---- K8: out = attn @ wo^T + bo (fp32) ------------------------------------
__global__ void pb_outproj_kernel(const f16* __restrict__ attn,
                                  const f16* __restrict__ woh,
                                  const float* __restrict__ bo,
                                  float* __restrict__ out) {
  __shared__ __align__(128) f16 sA[16 * E_DIM];   // 16 KB
  const int lane = threadIdx.x;
  const int m0 = blockIdx.x * 16;
  const int n0 = blockIdx.y * 64;

  stage_a_tile(sA, attn + (size_t)m0 * E_DIM, E_DIM, 16, E_DIM, lane);

  v8f acc[4];
  zero_acc(acc);
  gemm_16x64(sA, E_DIM, woh + (size_t)n0 * E_DIM, E_DIM, E_DIM, lane, acc);

  const int col = lane & 15, rsel = (lane >> 4) * 8;
#pragma unroll
  for (int t = 0; t < 4; ++t)
#pragma unroll
    for (int r = 0; r < 8; ++r) {
      int j = n0 + t * 16 + col;
      out[(size_t)(m0 + r + rsel) * E_DIM + j] = acc[t][r] + bo[j];
    }
}

// ---------------------------------------------------------------------------
extern "C" void kernel_launch(void* const* d_in, const int* in_sizes, int n_in,
                              void* d_out, int out_size, void* d_ws,
                              size_t ws_size, hipStream_t stream) {
  const float* query = (const float*)d_in[0];
  const float* pairs = (const float*)d_in[1];
  const float* ln_w  = (const float*)d_in[2];
  const float* ln_b  = (const float*)d_in[3];
  const float* wq    = (const float*)d_in[4];
  const float* wk    = (const float*)d_in[5];
  const float* wv    = (const float*)d_in[6];
  const float* wo    = (const float*)d_in[7];
  const float* bo    = (const float*)d_in[8];
  const float* lnp_w = (const float*)d_in[9];
  const float* lnp_b = (const float*)d_in[10];
  const float* wp    = (const float*)d_in[11];
  float* out = (float*)d_out;

  char* ws = (char*)d_ws;
  const size_t MB = 1024ull * 1024ull;
  f16* wqh   = (f16*)(ws + 0 * MB);          // 512 KB each
  f16* wkh   = (f16*)(ws + 512 * 1024ull);
  f16* wvh   = (f16*)(ws + 1 * MB);
  f16* woh   = (f16*)(ws + 1 * MB + 512 * 1024ull);
  f16* xh    = (f16*)(ws + 2 * MB);          // 2 MB
  f16* qh    = (f16*)(ws + 4 * MB);          // 2 MB
  f16* kh    = (f16*)(ws + 6 * MB);          // 2 MB
  f16* vh    = (f16*)(ws + 8 * MB);          // 2 MB
  f16* vT    = (f16*)(ws + 10 * MB);         // 2 MB
  f16* attn  = (f16*)(ws + 12 * MB);         // 2 MB
  f16* probs = (f16*)(ws + 16 * MB);         // 32 MB  [B,H,L,S] f16
  float* scores = (float*)(ws + 48 * MB);    // 64 MB  [B,H,L,S] f32

  const int NW = E_DIM * E_DIM;  // 262144 per weight matrix

  // K0: weights -> f16
  pb_cvt_f16_kernel<<<NW / 256, 256, 0, stream>>>(wq, wqh, NW);
  pb_cvt_f16_kernel<<<NW / 256, 256, 0, stream>>>(wk, wkh, NW);
  pb_cvt_f16_kernel<<<NW / 256, 256, 0, stream>>>(wv, wvh, NW);
  pb_cvt_f16_kernel<<<NW / 256, 256, 0, stream>>>(wo, woh, NW);

  // K1: LayerNorm(query)
  pb_ln_query_kernel<<<ROWS / 8, 256, 0, stream>>>(query, ln_w, ln_b, xh);

  // K2: q,k,v projections (q scaled by 1/sqrt(64))
  pb_qkv_gemm_kernel<<<dim3(ROWS / 16, E_DIM / 64, 3), 32, 0, stream>>>(
      xh, wqh, wkh, wvh, qh, kh, vh);

  // K3: pair bias into scores (single pass over 537 MB `pairs`)
  pb_pair_bias_kernel<<<(B_DIM * L_DIM * L_DIM) / 256, 256, 0, stream>>>(
      pairs, lnp_w, lnp_b, wp, scores);

  // K4: scores += q k^T
  pb_qk_kernel<<<dim3(L_DIM / 16, L_DIM / 64, B_DIM * H_NUM), 32, 0, stream>>>(
      qh, kh, scores);

  // K5: softmax rows
  pb_softmax_kernel<<<(B_DIM * H_NUM * L_DIM) / 8, 256, 0, stream>>>(scores,
                                                                     probs);

  // K6: v transpose for contiguous B-fragments
  pb_vtrans_kernel<<<(B_DIM * H_NUM * D_HEAD * L_DIM) / 256, 256, 0, stream>>>(
      vh, vT);

  // K7: attn = probs @ v
  pb_pv_kernel<<<dim3(L_DIM / 16, 1, B_DIM * H_NUM), 32, 0, stream>>>(probs, vT,
                                                                      attn);

  // K8: out = attn @ wo^T + bo
  pb_outproj_kernel<<<dim3(ROWS / 16, E_DIM / 64), 32, 0, stream>>>(attn, woh,
                                                                    bo, out);
}
